// Attention_7310034337899
// MI455X (gfx1250) — compile-verified, matrix operand order checked by reading, not run on hardware
//
#include <hip/hip_runtime.h>

// ---------------------------------------------------------------------------
// MI455X (gfx1250) attention block: bf16 WMMA everywhere.
// B=2 S=2048 D=4096 H=32 KV=8 HD=128. ~1 TFLOP of matrix math vs ~300MB of
// traffic -> matrix-compute bound -> all GEMMs + attention run on
// v_wmma_f32_16x16x32_bf16 with f32 accumulation. Softmax row reductions use
// DPP16 butterflies; LDS staging uses the CDNA5 async global->LDS path
// (ASYNCcnt-tracked global_load_async_to_lds_b128).
// ---------------------------------------------------------------------------

static constexpr int B_  = 2;
static constexpr int S_  = 2048;
static constexpr int D_  = 4096;
static constexpr int H_  = 32;
static constexpr int KV_ = 8;
static constexpr int HD_ = 128;
static constexpr int G_  = H_ / KV_;

typedef __bf16 bf16_t;
typedef __attribute__((ext_vector_type(16))) __bf16 bf16x16;
typedef __attribute__((ext_vector_type(8)))  __bf16 bf16x8;
typedef __attribute__((ext_vector_type(8)))  float  f32x8;
typedef __attribute__((ext_vector_type(4)))  float  f32x4;
typedef __attribute__((ext_vector_type(4)))  int    i32x4;

#if __has_builtin(__builtin_amdgcn_global_load_async_to_lds_b128)
#define HAVE_ASYNC_LDS 1
#else
#define HAVE_ASYNC_LDS 0
#endif

#if HAVE_ASYNC_LDS
typedef __attribute__((address_space(1))) i32x4* as1_i32x4p;
typedef __attribute__((address_space(3))) i32x4* as3_i32x4p;
__device__ inline void async_copy16(const bf16_t* g, bf16_t* l) {
  __builtin_amdgcn_global_load_async_to_lds_b128(
      (as1_i32x4p)(void*)g, (as3_i32x4p)(void*)l, 0, 0);
}
__device__ inline void async_wait0() {
#if __has_builtin(__builtin_amdgcn_s_wait_asynccnt)
  __builtin_amdgcn_s_wait_asynccnt(0);
#else
  asm volatile("s_wait_asynccnt 0" ::: "memory");
#endif
}
#endif

// ---------------------------------------------------------------------------
// WMMA helpers. 16x16x32 bf16, f32 accumulate.
// A-frag (16x32, row-major source): lane<16 -> row=lane, K = k0+{0..7,16..23};
// lane>=16 -> row=lane-16, K = k0+{8..15,24..31}. B-frag for C = A*W^T loads
// rows of W with the identical pattern (B[k][n] == W[n][k]).
// C-frag: col = lane&15, row = 8*(lane>>4) + vgpr.
// ---------------------------------------------------------------------------
__device__ inline f32x8 wmma_bf16(bf16x16 a, bf16x16 b, f32x8 c) {
  return __builtin_amdgcn_wmma_f32_16x16x32_bf16(
      /*neg_a=*/false, a, /*neg_b=*/false, b,
      /*c_mod=*/(short)0, c, /*reuse_a=*/false, /*reuse_b=*/false);
}

__device__ inline bf16x16 load_frag(const bf16_t* base, long ld, int k0) {
  unsigned lane = threadIdx.x & 31u;
  const bf16_t* p = base + (long)(lane & 15u) * ld + (k0 + (int)((lane >> 4) * 8u));
  bf16x8 lo = *reinterpret_cast<const bf16x8*>(p);
  bf16x8 hi = *reinterpret_cast<const bf16x8*>(p + 16);
  bf16x16 f;
#pragma unroll
  for (int i = 0; i < 8; ++i) { f[i] = lo[i]; f[8 + i] = hi[i]; }
  return f;
}

__device__ inline void store_c(float* p, float v)  { *p = v; }
__device__ inline void store_c(bf16_t* p, float v) { *p = (bf16_t)v; }

// ---------------------------------------------------------------------------
// DPP16 16-lane butterfly reductions (pure VALU). Tree: quad_perm xor1 (0xB1),
// quad_perm xor2 (0x4E), row_half_mirror (0x141), row_mirror (0x140).
// Result replicated across each 16-lane half of the wave32.
// ---------------------------------------------------------------------------
template <int CTRL>
__device__ inline float dpp_perm(float x) {
  return __int_as_float(__builtin_amdgcn_update_dpp(
      0, __float_as_int(x), CTRL, 0xF, 0xF, true));
}
__device__ inline float hmax16(float x) {
  x = fmaxf(x, dpp_perm<0xB1>(x));
  x = fmaxf(x, dpp_perm<0x4E>(x));
  x = fmaxf(x, dpp_perm<0x141>(x));
  x = fmaxf(x, dpp_perm<0x140>(x));
  return x;
}
__device__ inline float hsum16(float x) {
  x += dpp_perm<0xB1>(x);
  x += dpp_perm<0x4E>(x);
  x += dpp_perm<0x141>(x);
  x += dpp_perm<0x140>(x);
  return x;
}

// ---------------------------------------------------------------------------
// GEMM: C[M,N] = A[M,K] * W[N,K]^T, bf16 in, OutT out.
// Block tile 256x64, 256 threads = 8 waves; wave w owns rows [32w,32w+32)
// x 64 cols: 8 C frags, 8 WMMAs per k-step against one LDS-staged 64x32 W
// slab shared by all waves. Staging uses async global->LDS when available.
// ---------------------------------------------------------------------------
template <typename OutT>
__global__ __launch_bounds__(256) void gemm_nt(const bf16_t* __restrict__ A,
                                               const bf16_t* __restrict__ W,
                                               OutT* __restrict__ C,
                                               int M, int N, int K) {
  __shared__ bf16_t wt[64][40];  // 64 rows x 32 k (+8 pad); rows 80B apart (16B aligned)
  const int wid = threadIdx.x >> 5;
  const unsigned lane = threadIdx.x & 31u;
  const long row0 = (long)blockIdx.y * 256 + wid * 32;
  const long col0 = (long)blockIdx.x * 64;

  f32x8 acc[2][4] = {};
  const bf16_t* arow0 = A + row0 * (long)K;
  const bf16_t* arow1 = arow0 + (long)16 * K;

  const int t = (int)threadIdx.x;
  const int sr = t >> 2, sch = (t & 3) * 8;
  const bf16_t* wrow = W + (col0 + sr) * (long)K + sch;

  for (int k0 = 0; k0 < K; k0 += 32) {
    // Stage 64x32 W tile: 256 threads x 16B.
#if HAVE_ASYNC_LDS
    async_copy16(wrow + k0, &wt[sr][sch]);
    if (k0 + 32 < K) __builtin_prefetch(wrow + k0 + 32, 0, 1);
    async_wait0();
#else
    *reinterpret_cast<bf16x8*>(&wt[sr][sch]) =
        *reinterpret_cast<const bf16x8*>(wrow + k0);
    if (k0 + 32 < K) __builtin_prefetch(wrow + k0 + 32, 0, 1);
#endif
    __syncthreads();

    bf16x16 a0 = load_frag(arow0, K, k0);
    bf16x16 a1 = load_frag(arow1, K, k0);
#pragma unroll
    for (int j = 0; j < 4; ++j) {
      bf16x16 bfr = load_frag(&wt[j * 16][0], 40, 0);
      acc[0][j] = wmma_bf16(a0, bfr, acc[0][j]);
      acc[1][j] = wmma_bf16(a1, bfr, acc[1][j]);
    }
    __syncthreads();
  }

#pragma unroll
  for (int i = 0; i < 2; ++i)
#pragma unroll
    for (int j = 0; j < 4; ++j) {
      long col = col0 + j * 16 + (lane & 15u);
      long rbase = row0 + i * 16 + (long)(lane >> 4) * 8;
#pragma unroll
      for (int r = 0; r < 8; ++r) store_c(&C[(rbase + r) * N + col], acc[i][j][r]);
    }
}

// ---------------------------------------------------------------------------
// f32 -> bf16 conversion (one-shot, bandwidth-trivial), 4 elems/thread.
// ---------------------------------------------------------------------------
__global__ void cvt_kernel(const float* __restrict__ src, bf16_t* __restrict__ dst,
                           long n4) {
  long i = (long)blockIdx.x * blockDim.x + threadIdx.x;
  if (i >= n4) return;
  f32x4 v = *reinterpret_cast<const f32x4*>(src + i * 4);
  bf16_t o[4];
#pragma unroll
  for (int j = 0; j < 4; ++j) o[j] = (bf16_t)v[j];
  *reinterpret_cast<ulong1*>(dst + i * 4) = *reinterpret_cast<ulong1*>(o);
}

// ---------------------------------------------------------------------------
// RoPE: t[b,s,h,2k+j] = sum_m t[b,s,h,2k+m] * fc[s,k,j,m], in place on bf16.
// ---------------------------------------------------------------------------
__global__ void rope_kernel(bf16_t* __restrict__ t, const float* __restrict__ fc,
                            int nh, long total) {
  long i = (long)blockIdx.x * blockDim.x + threadIdx.x;
  if (i >= total) return;
  int k = (int)(i % (HD_ / 2));
  long tmp = i / (HD_ / 2);
  tmp /= nh;                       // drop head index
  int s = (int)(tmp % S_);
  long base = i * 2;               // layout [b,s,h,hd] => element index 2*i
  float t0 = (float)t[base], t1 = (float)t[base + 1];
  const float* f = fc + ((long)s * (HD_ / 2) + k) * 4;
  t[base]     = (bf16_t)(t0 * f[0] + t1 * f[1]);
  t[base + 1] = (bf16_t)(t0 * f[2] + t1 * f[3]);
}

// ---------------------------------------------------------------------------
// Flash attention (causal GQA). Grid: (S/64, B*H), 128 threads = 4 waves.
// Wave owns a 16-row query tile; key tiles of 32. All 8 K-frags of a tile are
// preloaded before the 8 QK WMMAs so the scheduler can issue partial
// s_wait_loadcnt instead of full stalls. Online softmax via DPP16; P bounced
// through per-wave LDS; V staged in per-wave LDS via async global->LDS.
// ---------------------------------------------------------------------------
__global__ __launch_bounds__(128) void attn_kernel(const bf16_t* __restrict__ Q,
                                                   const bf16_t* __restrict__ Km,
                                                   const bf16_t* __restrict__ V,
                                                   bf16_t* __restrict__ O) {
  __shared__ bf16_t p_lds[4][16][32];   // per-wave P tile (4KB)
  __shared__ bf16_t v_lds[4][32][128];  // per-wave V tile (32KB)

  const int wid = threadIdx.x >> 5;
  const unsigned lane = threadIdx.x & 31u;
  const int bh = blockIdx.y;
  const int b = bh / H_, h = bh % H_;
  const int kvh = h / G_;
  const int qrow0 = (int)blockIdx.x * 64 + wid * 16;

  const long q_ld = (long)H_ * HD_;
  const long kv_ld = (long)KV_ * HD_;
  const bf16_t* qbase = Q + ((b * (long)S_ + qrow0) * H_ + h) * HD_;
  const bf16_t* kb0 = Km + ((b * (long)S_) * KV_ + kvh) * HD_;
  const bf16_t* vb0 = V  + ((b * (long)S_) * KV_ + kvh) * HD_;

  bf16x16 qf[4];
#pragma unroll
  for (int c = 0; c < 4; ++c) qf[c] = load_frag(qbase, q_ld, c * 32);

  f32x8 o_acc[8] = {};
  float m_i[8], l_i[8];
#pragma unroll
  for (int r = 0; r < 8; ++r) { m_i[r] = -1e30f; l_i[r] = 0.f; }
  const float scale = 0.08838834764831845f;  // 1/sqrt(128)

  const int kend = qrow0 + 16;  // causal horizon for this wave's rows
  for (int kt = 0; kt < kend; kt += 32) {
    if (kt + 32 < kend) {  // pull next K/V tiles toward the caches
      __builtin_prefetch(kb0 + (long)(kt + 32 + (int)lane) * kv_ld, 0, 1);
      __builtin_prefetch(vb0 + (long)(kt + 32 + (int)lane) * kv_ld, 0, 1);
    }

    // ---- start async V-tile staging early (32 keys x 128; lane = key row)
#if HAVE_ASYNC_LDS
    {
      const bf16_t* vr = vb0 + (long)(kt + (int)lane) * kv_ld;
#pragma unroll
      for (int ch = 0; ch < 16; ++ch)
        async_copy16(vr + ch * 8, &v_lds[wid][lane][ch * 8]);
    }
#endif

    // ---- S = Q K^T : preload all 8 K-frags, then 8 WMMAs -----------------
    bf16x16 kf[2][4];
#pragma unroll
    for (int hf = 0; hf < 2; ++hf) {
      const bf16_t* kbase = kb0 + (long)(kt + hf * 16) * kv_ld;
#pragma unroll
      for (int c = 0; c < 4; ++c) kf[hf][c] = load_frag(kbase, kv_ld, c * 32);
    }
    f32x8 sc[2] = {};
#pragma unroll
    for (int hf = 0; hf < 2; ++hf)
#pragma unroll
      for (int c = 0; c < 4; ++c) sc[hf] = wmma_bf16(qf[c], kf[hf][c], sc[hf]);

    // ---- scale + causal mask (all-int compares) --------------------------
    const int row_g = qrow0 + (int)((lane >> 4) * 8u);
#pragma unroll
    for (int hf = 0; hf < 2; ++hf) {
      int col_g = kt + hf * 16 + (int)(lane & 15u);
#pragma unroll
      for (int r = 0; r < 8; ++r) {
        float x = sc[hf][r] * scale;
        sc[hf][r] = (col_g <= row_g + r) ? x : -1e9f;
      }
    }
    // ---- online softmax: DPP16 row reductions ----------------------------
    float alpha[8];
#pragma unroll
    for (int r = 0; r < 8; ++r) {
      float mx = hmax16(fmaxf(sc[0][r], sc[1][r]));
      float mn = fmaxf(m_i[r], mx);
      alpha[r] = __expf(m_i[r] - mn);
      m_i[r] = mn;
    }
#pragma unroll
    for (int hf = 0; hf < 2; ++hf)
#pragma unroll
      for (int r = 0; r < 8; ++r) sc[hf][r] = __expf(sc[hf][r] - m_i[r]);
#pragma unroll
    for (int r = 0; r < 8; ++r) {
      float rs = hsum16(sc[0][r] + sc[1][r]);
      l_i[r] = l_i[r] * alpha[r] + rs;
    }
#pragma unroll
    for (int c = 0; c < 8; ++c)
#pragma unroll
      for (int r = 0; r < 8; ++r) o_acc[c][r] *= alpha[r];

    // ---- P: C-layout -> LDS -> A-layout ----------------------------------
#pragma unroll
    for (int hf = 0; hf < 2; ++hf)
#pragma unroll
      for (int r = 0; r < 8; ++r)
        p_lds[wid][(lane >> 4) * 8 + r][hf * 16 + (lane & 15u)] = (bf16_t)sc[hf][r];

    // ---- finish V-tile staging -------------------------------------------
#if HAVE_ASYNC_LDS
    async_wait0();
#else
    {
      const bf16_t* vr = vb0 + (long)(kt + (int)lane) * kv_ld;
#pragma unroll
      for (int ch = 0; ch < 16; ++ch)
        *reinterpret_cast<bf16x8*>(&v_lds[wid][lane][ch * 8]) =
            *reinterpret_cast<const bf16x8*>(vr + ch * 8);
    }
#endif
    asm volatile("s_wait_dscnt 0" ::: "memory");  // wave-private LDS RAW fence

    // ---- O += P V : 8 hd-chunks ------------------------------------------
    bf16x16 pf = load_frag(&p_lds[wid][0][0], 32, 0);
    const unsigned nloc = lane & 15u;
    const int ksh = (int)((lane >> 4) * 8u);
#pragma unroll
    for (int c = 0; c < 8; ++c) {
      bf16x16 vf;
#pragma unroll
      for (int j = 0; j < 8; ++j) {
        vf[j]     = v_lds[wid][ksh + j][c * 16 + nloc];
        vf[8 + j] = v_lds[wid][ksh + 16 + j][c * 16 + nloc];
      }
      o_acc[c] = wmma_bf16(pf, vf, o_acc[c]);
    }
  }

  // ---- epilogue: normalize, store [b,s,h,hd] ------------------------------
#pragma unroll
  for (int c = 0; c < 8; ++c) {
    int col = c * 16 + (int)(lane & 15u);
#pragma unroll
    for (int r = 0; r < 8; ++r) {
      long row = qrow0 + (long)(lane >> 4) * 8 + r;
      float v = o_acc[c][r] / l_i[r];
      O[((b * (long)S_ + row) * H_ + h) * HD_ + col] = (bf16_t)v;
    }
  }
}

// ---------------------------------------------------------------------------
// Host orchestration.
// ---------------------------------------------------------------------------
extern "C" void kernel_launch(void* const* d_in, const int* in_sizes, int n_in,
                              void* d_out, int out_size, void* d_ws, size_t ws_size,
                              hipStream_t stream) {
  (void)in_sizes; (void)n_in; (void)out_size; (void)ws_size;
  const float* x  = (const float*)d_in[0];
  const float* fc = (const float*)d_in[1];
  const float* wq = (const float*)d_in[2];
  const float* wk = (const float*)d_in[3];
  const float* wv = (const float*)d_in[4];
  const float* wo = (const float*)d_in[5];
  float* out = (float*)d_out;

  const long NX  = (long)B_ * S_ * D_;          // x / attn-out elems
  const long NWQ = (long)H_ * HD_ * D_;         // wq / wo elems
  const long NWK = (long)KV_ * HD_ * D_;        // wk / wv elems
  const long NQ  = (long)B_ * S_ * H_ * HD_;
  const long NK  = (long)B_ * S_ * KV_ * HD_;

  char* ws = (char*)d_ws;
  size_t off = 0;
  auto take = [&](long elems) {
    bf16_t* p = (bf16_t*)(ws + off);
    off += ((size_t)elems * 2 + 255) & ~(size_t)255;
    return p;
  };
  bf16_t* xb  = take(NX);
  bf16_t* wqb = take(NWQ);
  bf16_t* wkb = take(NWK);
  bf16_t* wvb = take(NWK);
  bf16_t* wob = take(NWQ);
  bf16_t* qb  = take(NQ);
  bf16_t* kb  = take(NK);
  bf16_t* vb  = take(NK);
  bf16_t* ob  = take(NQ);

  auto cvt = [&](const float* s, bf16_t* d, long n) {
    long n4 = n / 4;
    cvt_kernel<<<dim3((unsigned)((n4 + 255) / 256)), 256, 0, stream>>>(s, d, n4);
  };
  cvt(x, xb, NX);
  cvt(wq, wqb, NWQ);
  cvt(wk, wkb, NWK);
  cvt(wv, wvb, NWK);
  cvt(wo, wob, NWQ);

  const int M = B_ * S_;  // 4096
  gemm_nt<bf16_t><<<dim3((H_ * HD_) / 64, M / 256), 256, 0, stream>>>(xb, wqb, qb, M, H_ * HD_, D_);
  gemm_nt<bf16_t><<<dim3((KV_ * HD_) / 64, M / 256), 256, 0, stream>>>(xb, wkb, kb, M, KV_ * HD_, D_);
  gemm_nt<bf16_t><<<dim3((KV_ * HD_) / 64, M / 256), 256, 0, stream>>>(xb, wvb, vb, M, KV_ * HD_, D_);

  const long nrq = (long)B_ * S_ * H_ * (HD_ / 2);
  const long nrk = (long)B_ * S_ * KV_ * (HD_ / 2);
  rope_kernel<<<dim3((unsigned)((nrq + 255) / 256)), 256, 0, stream>>>(qb, fc, H_, nrq);
  rope_kernel<<<dim3((unsigned)((nrk + 255) / 256)), 256, 0, stream>>>(kb, fc, KV_, nrk);

  attn_kernel<<<dim3(S_ / 64, B_ * H_), 128, 0, stream>>>(qb, kb, vb, ob);

  gemm_nt<float><<<dim3(D_ / 64, M / 256), 256, 0, stream>>>(ob, wob, out, M, D_, H_ * HD_);
}
